// TransformerBlock_10230612099646
// MI455X (gfx1250) — matrix operand order repaired and checked by reading, not compiled, not run on hardware
//
#include <hip/hip_runtime.h>
#include <hip/hip_bf16.h>
#include <math.h>

// ---------------- problem constants ----------------
#define B_   8
#define S_   1024
#define D_   768
#define H_   12
#define HD_  64
#define MLP_ 3072
#define M_   (B_*S_)     // 8192 rows

typedef __attribute__((ext_vector_type(16))) _Float16 v16h;
typedef __attribute__((ext_vector_type(8)))  float    v8f;

union Frag { v16h v; unsigned u[8]; };

// LDS offset of a __shared__ pointer: shared aperture base has zero low 32 bits.
__device__ __forceinline__ unsigned lds_off(const void* p) {
  return (unsigned)(uintptr_t)p;
}

// async copy of 16 bytes global -> LDS (per lane), tracked by ASYNCcnt
__device__ __forceinline__ void async_b128(unsigned lds, const void* gptr) {
  asm volatile("global_load_async_to_lds_b128 %0, %1, off"
               :: "v"(lds), "v"(gptr) : "memory");
}
__device__ __forceinline__ void wait_async0() {
  asm volatile("s_wait_asynccnt 0" ::: "memory");
}

// ---------------- instance-norm stats (per batch over S*D) ----------------
__global__ __launch_bounds__(256) void k_instnorm_stats(const float* __restrict__ x,
                                                        float* __restrict__ st) {
  const int NB = S_ * D_;
  const int b = blockIdx.x, tid = threadIdx.x;
  const float4* p = (const float4*)(x + (size_t)b * NB);
  float s = 0.f, s2 = 0.f;
  for (int i = tid; i < NB / 4; i += 256) {
    float4 v = p[i];
    s  += v.x + v.y + v.z + v.w;
    s2 += v.x*v.x + v.y*v.y + v.z*v.z + v.w*v.w;
  }
  __shared__ float rs[256], rq[256];
  rs[tid] = s; rq[tid] = s2; __syncthreads();
  for (int off = 128; off > 0; off >>= 1) {
    if (tid < off) { rs[tid] += rs[tid+off]; rq[tid] += rq[tid+off]; }
    __syncthreads();
  }
  if (tid == 0) {
    float mu  = rs[0] / NB;
    float var = rq[0] / NB - mu * mu;
    st[2*b]   = mu;
    st[2*b+1] = rsqrtf(var + 1e-5f);
  }
}

// ---------------- normalize + f32 -> f16 ----------------
struct alignas(8) h4 { _Float16 a, b, c, d; };

__global__ __launch_bounds__(256) void k_normalize_f16(const float* __restrict__ x,
                                                       const float* __restrict__ st,
                                                       _Float16* __restrict__ o) {
  int idx = (blockIdx.x * 256 + threadIdx.x) * 4;
  int b = idx / (S_ * D_);
  float mu = st[2*b], rstd = st[2*b+1];
  float4 v = *(const float4*)(x + idx);
  h4 r;
  r.a = (_Float16)((v.x - mu) * rstd);
  r.b = (_Float16)((v.y - mu) * rstd);
  r.c = (_Float16)((v.z - mu) * rstd);
  r.d = (_Float16)((v.w - mu) * rstd);
  *(h4*)(o + idx) = r;
}

// ---------------- weight transpose + f32 -> f16 : w[K][N] -> wt[N][K] ----------------
__global__ __launch_bounds__(256) void k_transpose_f16(const float* __restrict__ w,
                                                       _Float16* __restrict__ wt,
                                                       int K, int N) {
  int idx = blockIdx.x * 256 + threadIdx.x;
  if (idx >= K * N) return;
  int k = idx / N, n = idx % N;
  wt[(size_t)n * K + k] = (_Float16)w[idx];
}

// ---------------- WMMA GEMM: C[M,N] = A[M,K] * Bt[N,K]^T, fused epilogues ----------------
#define BM 128
#define BN 128
#define BK 32
#define LDT 40   // padded LDS row stride in halves (80B, 16B-aligned rows)

enum { EPI_QKV = 0, EPI_RES = 1, EPI_GELU = 2 };

__global__ __launch_bounds__(256)
void k_gemm(const _Float16* __restrict__ A, const _Float16* __restrict__ Bt,
            int M, int N, int K, int epi,
            const float* __restrict__ bias, const float* __restrict__ resid,
            float* __restrict__ outf, _Float16* __restrict__ outh,
            _Float16* __restrict__ qO, _Float16* __restrict__ kO,
            _Float16* __restrict__ vtO) {
  __shared__ _Float16 As[2][BM * LDT];   // double-buffered: async pipeline
  __shared__ _Float16 Bs[2][BN * LDT];

  const int tid = threadIdx.x;
  const int lane = tid & 31, lHi = lane >> 4, l16 = lane & 15;
  const int w  = tid >> 5;
  const int wm = w & 1, wn = w >> 1;            // 2x4 wave grid -> 64x32 per wave
  const int row0 = blockIdx.y * BM, col0 = blockIdx.x * BN;

  // this thread's two 8-half chunks of each tile
  const int r_ld  = tid >> 2;                    // rows 0..63 (+64 on 2nd chunk)
  const int c8_ld = (tid & 3) << 3;              // 0,8,16,24

  auto issue_tile = [&](int buf, int k0) {
#pragma unroll
    for (int it = 0; it < 2; ++it) {
      int r = r_ld + it * 64;
      async_b128(lds_off(&As[buf][r * LDT + c8_ld]),
                 A  + (size_t)(row0 + r) * K + k0 + c8_ld);
      async_b128(lds_off(&Bs[buf][r * LDT + c8_ld]),
                 Bt + (size_t)(col0 + r) * K + k0 + c8_ld);
    }
  };

  v8f zero = {0.f,0.f,0.f,0.f,0.f,0.f,0.f,0.f};
  v8f acc[4][2];
#pragma unroll
  for (int i = 0; i < 4; ++i)
#pragma unroll
    for (int j = 0; j < 2; ++j) acc[i][j] = zero;

  const int nk = K / BK;
  issue_tile(0, 0);

  for (int kt = 0; kt < nk; ++kt) {
    const int cur = kt & 1;
    wait_async0();          // our slice of tile kt landed
    __syncthreads();        // everyone's slice landed; prev compute done
    if (kt + 1 < nk) issue_tile(cur ^ 1, (kt + 1) * BK);  // overlaps WMMA below

    Frag af[4], bf[2];
#pragma unroll
    for (int mt = 0; mt < 4; ++mt) {
      const unsigned* rp = (const unsigned*)(&As[cur][(wm*64 + mt*16 + l16) * LDT]);
#pragma unroll
      for (int d = 0; d < 8; ++d) {
        int kk = (d < 4) ? (8*lHi + 2*d) : (16 + 8*lHi + 2*(d - 4));
        af[mt].u[d] = rp[kk >> 1];
      }
    }
#pragma unroll
    for (int nt = 0; nt < 2; ++nt) {
      const unsigned* rp = (const unsigned*)(&Bs[cur][(wn*32 + nt*16 + l16) * LDT]);
#pragma unroll
      for (int d = 0; d < 8; ++d) bf[nt].u[d] = rp[(d < 4) ? (4*lHi + d) : (4 + 4*lHi + d)];
    }
#pragma unroll
    for (int mt = 0; mt < 4; ++mt)
#pragma unroll
      for (int nt = 0; nt < 2; ++nt)
        acc[mt][nt] = __builtin_amdgcn_wmma_f32_16x16x32_f16(
            false, af[mt].v, false, bf[nt].v, (short)0, acc[mt][nt], false, false);
  }

  // fused epilogue
#pragma unroll
  for (int mt = 0; mt < 4; ++mt)
    for (int nt = 0; nt < 2; ++nt)
#pragma unroll
      for (int r = 0; r < 8; ++r) {
        int grow = row0 + wm*64 + mt*16 + 8*lHi + r;
        int gcol = col0 + wn*32 + nt*16 + l16;
        float v = acc[mt][nt][r];
        if (epi == EPI_QKV) {
          int t = gcol / D_, rem = gcol % D_;
          int hh = rem >> 6, hd = rem & 63;
          int bb = grow >> 10, s = grow & 1023;
          size_t qk = (((size_t)(bb*H_ + hh) * S_ + s) << 6) + hd;
          if (t == 0)      qO[qk] = (_Float16)(v * 0.125f);   // fold 1/sqrt(HD)
          else if (t == 1) kO[qk] = (_Float16)v;
          else             vtO[(((size_t)(bb*H_ + hh) * HD_ + hd) << 10) + s] = (_Float16)v;
        } else if (epi == EPI_RES) {
          size_t o = (size_t)grow * N + gcol;
          outf[o] = v + bias[gcol] + resid[o];
        } else { // EPI_GELU -> f16
          float t = v + bias[gcol];
          float g = 0.5f * t * (1.f + erff(t * 0.70710678118f));
          outh[(size_t)grow * N + gcol] = (_Float16)g;
        }
      }
}

// ---------------- flash attention: per (b,h) x 64 query rows ----------------
__global__ __launch_bounds__(128)
void k_attn(const _Float16* __restrict__ q, const _Float16* __restrict__ kmat,
            const _Float16* __restrict__ vt, _Float16* __restrict__ o) {
  __shared__ _Float16 Qs[64 * 72];
  __shared__ _Float16 Ks[64 * 72];
  __shared__ _Float16 Vs[64 * 72];
  __shared__ _Float16 Ps[64 * 72];
  __shared__ float    Ss[64 * 68];
  __shared__ float    rowM[64], rowL[64], rowC[64];

  const int tid = threadIdx.x;
  const int lane = tid & 31, lHi = lane >> 4, l16 = lane & 15;
  const int w = tid >> 5;                          // wave 0..3 -> hd tile
  const int bh = blockIdx.y;
  const int q0 = blockIdx.x * 64;
  const size_t headBase = (size_t)bh * S_ * HD_;   // q,k: [bh][s][hd]
  const size_t vBase    = (size_t)bh * HD_ * S_;   // vt:  [bh][hd][s]

  const int r_ld  = tid >> 3;                      // rows 0..15 (+16/32/48)
  const int c8_ld = (tid & 7) << 3;                // 0..56

#pragma unroll
  for (int it = 0; it < 4; ++it) {                 // load 64x64 Q tile (once)
    int r = r_ld + it * 16;
    uint4 d = *(const uint4*)(q + headBase + (size_t)(q0 + r) * HD_ + c8_ld);
    *(uint4*)(Qs + r * 72 + c8_ld) = d;
  }
  if (tid < 64) { rowM[tid] = -3.0e38f; rowL[tid] = 0.f; }

  v8f zero = {0.f,0.f,0.f,0.f,0.f,0.f,0.f,0.f};
  v8f oacc[4];
#pragma unroll
  for (int mt = 0; mt < 4; ++mt) oacc[mt] = zero;

  for (int sc = 0; sc < S_; sc += 64) {
    // async: K chunk [64 s][64 hd] and V chunk [64 hd][64 s] land in parallel
#pragma unroll
    for (int it = 0; it < 4; ++it) {
      int r = r_ld + it * 16;
      async_b128(lds_off(&Ks[r * 72 + c8_ld]),
                 kmat + headBase + (size_t)(sc + r) * HD_ + c8_ld);
      async_b128(lds_off(&Vs[r * 72 + c8_ld]),
                 vt + vBase + (size_t)r * S_ + sc + c8_ld);
    }
    wait_async0();
    __syncthreads();

    // scores: wave w owns score-column tile w (16 cols), 4 row tiles, K=64 (2 steps)
#pragma unroll
    for (int mt = 0; mt < 4; ++mt) {
      v8f sacc = zero;
#pragma unroll
      for (int ks = 0; ks < 2; ++ks) {
        Frag af, bf;
        const unsigned* ra = (const unsigned*)(Qs + (mt*16 + l16) * 72);
        const unsigned* rb = (const unsigned*)(Ks + (w*16  + l16) * 72);
#pragma unroll
        for (int d = 0; d < 8; ++d) {
          int kk = (d < 4) ? (8*lHi + 2*d) : (16 + 8*lHi + 2*(d - 4));
          af.u[d] = ra[(ks*32 + kk) >> 1];
          bf.u[d] = rb[ks*16 + ((d < 4) ? (4*lHi + d) : (4 + 4*lHi + d))];
        }
        sacc = __builtin_amdgcn_wmma_f32_16x16x32_f16(
            false, af.v, false, bf.v, (short)0, sacc, false, false);
      }
#pragma unroll
      for (int r = 0; r < 8; ++r)
        Ss[(mt*16 + 8*lHi + r) * 68 + w*16 + l16] = sacc[r];
    }
    __syncthreads();

    // online softmax per row
    if (tid < 64) {
      int r = tid;
      float mloc = -3.0e38f;
      for (int j = 0; j < 64; ++j) mloc = fmaxf(mloc, Ss[r*68 + j]);
      float nm   = fmaxf(rowM[r], mloc);
      float corr = __expf(rowM[r] - nm);
      float l    = rowL[r] * corr;
      for (int j = 0; j < 64; ++j) {
        float p = __expf(Ss[r*68 + j] - nm);
        l += p;
        Ps[r*72 + j] = (_Float16)p;
      }
      rowM[r] = nm; rowL[r] = l; rowC[r] = corr;
    }
    __syncthreads();

    // rescale O accumulators, then O += P @ V
#pragma unroll
    for (int mt = 0; mt < 4; ++mt)
#pragma unroll
      for (int r = 0; r < 8; ++r)
        oacc[mt][r] = oacc[mt][r] * rowC[mt*16 + 8*lHi + r];

#pragma unroll
    for (int mt = 0; mt < 4; ++mt) {
#pragma unroll
      for (int ks = 0; ks < 2; ++ks) {
        Frag af, bf;
        const unsigned* ra = (const unsigned*)(Ps + (mt*16 + l16) * 72);
        const unsigned* rb = (const unsigned*)(Vs + (w*16  + l16) * 72);
#pragma unroll
        for (int d = 0; d < 8; ++d) {
          int kk = (d < 4) ? (8*lHi + 2*d) : (16 + 8*lHi + 2*(d - 4));
          af.u[d] = ra[(ks*32 + kk) >> 1];
          bf.u[d] = rb[ks*16 + ((d < 4) ? (4*lHi + d) : (4 + 4*lHi + d))];
        }
        oacc[mt] = __builtin_amdgcn_wmma_f32_16x16x32_f16(
            false, af.v, false, bf.v, (short)0, oacc[mt], false, false);
      }
    }
    __syncthreads();   // before next chunk's async overwrites Ks/Vs
  }

  const int bb = bh / H_, hh = bh % H_;
#pragma unroll
  for (int mt = 0; mt < 4; ++mt)
#pragma unroll
    for (int r = 0; r < 8; ++r) {
      int s = q0 + mt*16 + 8*lHi + r;
      float val = oacc[mt][r] / rowL[mt*16 + 8*lHi + r];
      o[(size_t)(bb*S_ + s) * D_ + hh*HD_ + w*16 + l16] = (_Float16)val;
    }
}

// ---------------- host orchestration ----------------
extern "C" void kernel_launch(void* const* d_in, const int* in_sizes, int n_in,
                              void* d_out, int out_size, void* d_ws, size_t ws_size,
                              hipStream_t stream) {
  const float* x     = (const float*)d_in[0];
  const float* w_qkv = (const float*)d_in[1];
  const float* w_out = (const float*)d_in[2];
  const float* b_out = (const float*)d_in[3];
  const float* w1    = (const float*)d_in[4];
  const float* b1    = (const float*)d_in[5];
  const float* w2    = (const float*)d_in[6];
  const float* b2    = (const float*)d_in[7];
  float* out = (float*)d_out;

  char* ws = (char*)d_ws;
  size_t off = 0;
  auto alloc = [&](size_t bytes) { void* p = ws + off; off += (bytes + 255) & ~(size_t)255; return p; };

  float*     st    = (float*)    alloc(4 * B_ * sizeof(float));       // st1 | st2
  _Float16*  xn1   = (_Float16*) alloc((size_t)M_ * D_ * 2);
  _Float16*  wqkvT = (_Float16*) alloc((size_t)3 * D_ * D_ * 2);
  _Float16*  woutT = (_Float16*) alloc((size_t)D_ * D_ * 2);
  _Float16*  w1T   = (_Float16*) alloc((size_t)MLP_ * D_ * 2);
  _Float16*  w2T   = (_Float16*) alloc((size_t)D_ * MLP_ * 2);
  _Float16*  qb    = (_Float16*) alloc((size_t)M_ * D_ * 2);
  _Float16*  kb    = (_Float16*) alloc((size_t)M_ * D_ * 2);
  _Float16*  vtb   = (_Float16*) alloc((size_t)M_ * D_ * 2);
  _Float16*  ao    = (_Float16*) alloc((size_t)M_ * D_ * 2);
  float*     x1    = (float*)    alloc((size_t)M_ * D_ * 4);
  _Float16*  xn2   = (_Float16*) alloc((size_t)M_ * D_ * 2);
  _Float16*  m1    = (_Float16*) alloc((size_t)M_ * MLP_ * 2);

  // norm1
  k_instnorm_stats<<<B_, 256, 0, stream>>>(x, st);
  k_normalize_f16<<<(M_ * D_ / 4) / 256, 256, 0, stream>>>(x, st, xn1);
  // weight prep (f16, [N][K])
  k_transpose_f16<<<(D_*3*D_ + 255)/256, 256, 0, stream>>>(w_qkv, wqkvT, D_, 3*D_);
  k_transpose_f16<<<(D_*D_   + 255)/256, 256, 0, stream>>>(w_out, woutT, D_, D_);
  k_transpose_f16<<<(D_*MLP_ + 255)/256, 256, 0, stream>>>(w1,    w1T,   D_, MLP_);
  k_transpose_f16<<<(MLP_*D_ + 255)/256, 256, 0, stream>>>(w2,    w2T,   MLP_, D_);
  // QKV projection -> q (scaled), k, v^T
  dim3 g1(3*D_/BN, M_/BM);
  k_gemm<<<g1, 256, 0, stream>>>(xn1, wqkvT, M_, 3*D_, D_, EPI_QKV,
                                 nullptr, nullptr, nullptr, nullptr, qb, kb, vtb);
  // attention
  dim3 ga(S_/64, B_*H_);
  k_attn<<<ga, 128, 0, stream>>>(qb, kb, vtb, ao);
  // out projection + bias + residual -> x1 (f32)
  dim3 g2(D_/BN, M_/BM);
  k_gemm<<<g2, 256, 0, stream>>>(ao, woutT, M_, D_, D_, EPI_RES,
                                 b_out, x, x1, nullptr, nullptr, nullptr, nullptr);
  // norm2
  k_instnorm_stats<<<B_, 256, 0, stream>>>(x1, st + 2*B_);
  k_normalize_f16<<<(M_ * D_ / 4) / 256, 256, 0, stream>>>(x1, st + 2*B_, xn2);
  // MLP up + GELU -> f16
  dim3 g3(MLP_/BN, M_/BM);
  k_gemm<<<g3, 256, 0, stream>>>(xn2, w1T, M_, MLP_, D_, EPI_GELU,
                                 b1, nullptr, nullptr, m1, nullptr, nullptr, nullptr);
  // MLP down + bias + residual -> out (f32)
  dim3 g4(D_/BN, M_/BM);
  k_gemm<<<g4, 256, 0, stream>>>(m1, w2T, M_, D_, MLP_, EPI_RES,
                                 b2, x1, out, nullptr, nullptr, nullptr, nullptr);
  (void)in_sizes; (void)n_in; (void)out_size; (void)ws_size;
}
